// NLSApproxTime_73753178407223
// MI455X (gfx1250) — matrix operand order repaired and checked by reading, not compiled, not run on hardware
//
#include <hip/hip_runtime.h>

// ---------------- problem constants (from reference) ----------------
#define T_ 3
#define C_ 64
#define H_ 192
#define W_ 192
#define PS_ 7
#define PAD_ 3
#define KK_ 7           // top-K
#define STRIDE0_ 4
#define NQ_ 48          // NQH == NQW
#define Q_ (T_*NQ_*NQ_) // 6912
#define S_ 2
#define QS_ (Q_*S_)     // 13824
#define R_ 9
#define FEAT_ (PS_*PS_*C_)   // 3136
#define NEG_INF (-3.402823e38f)

typedef __attribute__((ext_vector_type(16))) __bf16 v16bf;
typedef __attribute__((ext_vector_type(8)))  float  v8f;
typedef __attribute__((ext_vector_type(4)))  unsigned int u32x4;

union ABPack { v16bf v; u32x4 q[2]; };

__device__ __forceinline__ int reflect_i(int i, int n) {
  i = (i < 0) ? -i : i;
  return (i >= n) ? (2*n - 2 - i) : i;
}

#define WMMA_BF16(A, B, CACC) \
  __builtin_amdgcn_wmma_f32_16x16x32_bf16(false, (A), false, (B), (short)0, (CACC), false, false)

// ---------------- 0) f32 (T,C,H,W) -> bf16 channels-last (T,H,W,C) ----------------
extern "C" __global__ void nls_convert_kernel(const float* __restrict__ src,
                                              unsigned short* __restrict__ dst) {
  size_t i = (size_t)blockIdx.x * blockDim.x + threadIdx.x;
  size_t N = (size_t)T_ * C_ * H_ * W_;
  if (i >= N) return;
  int w = (int)(i % W_); size_t r1 = i / W_;
  int h = (int)(r1 % H_); size_t r2 = r1 / H_;
  int c = (int)(r2 % C_); int t = (int)(r2 / C_);
  unsigned int u = __float_as_uint(src[i]);
  unsigned int b = (u + 0x7FFFu + ((u >> 16) & 1u)) >> 16;  // RNE f32->bf16
  dst[(((size_t)t * H_ + h) * W_ + w) * C_ + c] = (unsigned short)b;
}

// ---------------- 1) exact search: wave32 per 4x4 query group ----------------
// WMMA diagonal mapping: A = 16 query feature chunks (from LDS tile of vid0),
// B = same 16 queries' vid1 chunk at fixed delta; diag(D) = 16 scores.
extern "C" __global__ void __launch_bounds__(32)
nls_exact_kernel(const unsigned short* __restrict__ v0,
                 const unsigned short* __restrict__ v1,
                 float* __restrict__ dists, int* __restrict__ inds) {
  const int g  = blockIdx.x;          // T_*12*12 groups
  const int t  = g / 144;
  const int gh = (g % 144) / 12;
  const int gw = g % 12;
  const int lane = threadIdx.x;
  const int r  = lane & 15;           // A row / D column owned-query
  const bool hi = lane >= 16;

  __shared__ __align__(16) unsigned short tile0[19 * 19 * C_];  // 4x4 queries + 7x7 patch
  __shared__ __align__(16) float diag[256];

  const int h0 = gh * 16 - PAD_;
  const int w0 = gw * 16 - PAD_;
  for (int rec = lane; rec < 19 * 19; rec += 32) {
    int ti = rec / 19, tj = rec % 19;
    int h = reflect_i(h0 + ti, H_), w = reflect_i(w0 + tj, W_);
    const u32x4* s = (const u32x4*)(v0 + (((size_t)t * H_ + h) * W_ + w) * C_);
    u32x4* d = (u32x4*)(tile0 + rec * C_);
#pragma unroll
    for (int p = 0; p < 8; ++p) d[p] = s[p];
  }
  __syncthreads();

  const int qi = r / 4, qj = r % 4;
  const int hqr = (gh * 4 + qi) * STRIDE0_;
  const int wqr = (gw * 4 + qj) * STRIDE0_;

  float best[KK_]; int bidx[KK_];
#pragma unroll
  for (int i = 0; i < KK_; ++i) { best[i] = NEG_INF; bidx[i] = 0; }

  const int o0 = hi ? 16 : 0, o1 = hi ? 24 : 8;   // A k-group offsets (bf16 elems)
  const int bo = hi ? 16 : 0;                     // B k offset
  const unsigned short* base1 = v1 + (size_t)t * H_ * W_ * C_;
  const unsigned short* atile = tile0 + ((qi * 4) * 19 + qj * 4) * C_;

  for (int d = 0; d < 64; ++d) {
    const int dy = d / 8 - 4, dx = d % 8 - 4;
    // hoist reflected column offsets (per-delta, per-lane constants)
    int coloff[PS_];
#pragma unroll
    for (int p = 0; p < PS_; ++p)
      coloff[p] = reflect_i(wqr + p - PAD_ + dx, W_) * C_;

    v8f acc0 = {}, acc1 = {};
    for (int py = 0; py < PS_; ++py) {
      const unsigned short* brow =
          base1 + (size_t)reflect_i(hqr + py - PAD_ + dy, H_) * (W_ * C_);
      const unsigned short* arow = atile + (py * 19) * C_;
#pragma unroll
      for (int px = 0; px < PS_; ++px) {
        ABPack a0, b0, a1, b1;
        const unsigned short* ap = arow + px * C_;       // LDS, immediate offsets
        const unsigned short* bp = brow + coloff[px];    // global, L2-resident
        a0.q[0] = *(const u32x4*)(ap + o0);
        a0.q[1] = *(const u32x4*)(ap + o1);
        b0.q[0] = ((const u32x4*)(bp + bo))[0];
        b0.q[1] = ((const u32x4*)(bp + bo))[1];
        acc0 = WMMA_BF16(a0.v, b0.v, acc0);
        a1.q[0] = *(const u32x4*)(ap + 32 + o0);
        a1.q[1] = *(const u32x4*)(ap + 32 + o1);
        b1.q[0] = ((const u32x4*)(bp + 32 + bo))[0];
        b1.q[1] = ((const u32x4*)(bp + 32 + bo))[1];
        acc1 = WMMA_BF16(a1.v, b1.v, acc1);
      }
    }
    // extract diagonal through LDS
#pragma unroll
    for (int v = 0; v < 8; ++v) diag[lane * 8 + v] = acc0[v] + acc1[v];
    __syncthreads();
    if (lane < 16) {
      float cs = (r < 8) ? diag[r * 8 + r] : diag[(16 + r) * 8 + (r - 8)];
      int cd = d;
      // branchless stable bubble insert (ties keep earlier delta -> matches top_k)
#pragma unroll
      for (int i = 0; i < KK_; ++i) {
        const bool gt = cs > best[i];
        const float lf = gt ? best[i] : cs;
        const int   ld = gt ? bidx[i] : cd;
        best[i] = gt ? cs : best[i];
        bidx[i] = gt ? cd : bidx[i];
        cs = lf; cd = ld;
      }
    }
    __syncthreads();
  }

  if (lane < 16) {
    const int qidx = (t * NQ_ + gh * 4 + qi) * NQ_ + gw * 4 + qj;
#pragma unroll
    for (int i = 0; i < KK_; ++i) {
      dists[qidx * KK_ + i] = best[i];
      const int d = bidx[i];
      const int dy = d / 8 - 4, dx = d % 8 - 4;
      const int b = (qidx * KK_ + i) * 3;
      inds[b + 0] = t;
      inds[b + 1] = reflect_i(hqr + dy, H_);
      inds[b + 2] = reflect_i(wqr + dx, W_);
    }
  }
}

// ---------------- 2) flow-shift top-K indices to +/-1 frame ----------------
extern "C" __global__ void nls_temporal_kernel(const int* __restrict__ inds_e,
                                               const float* __restrict__ fflow,
                                               const float* __restrict__ bflow,
                                               int* __restrict__ inds_t) {
  int i = blockIdx.x * blockDim.x + threadIdx.x;   // over Q_*KK_
  if (i >= Q_ * KK_) return;
  const int q = i / KK_, k = i % KK_;
  const int b = (q * KK_ + k) * 3;
  const int t = inds_e[b], h = inds_e[b + 1], w = inds_e[b + 2];
#pragma unroll
  for (int s = 0; s < S_; ++s) {
    const float* fl = (s == 0) ? fflow : bflow;
    const int st = (s == 0) ? 1 : -1;
    const float fx = fl[(((size_t)t * 2 + 0) * H_ + h) * W_ + w];
    const float fy = fl[(((size_t)t * 2 + 1) * H_ + h) * W_ + w];
    const int dx = (int)rintf(fx), dy = (int)rintf(fy);
    int tt = t + st; tt = tt < 0 ? 0 : (tt > T_ - 1 ? T_ - 1 : tt);
    const int o = ((q * S_ + s) * KK_ + k) * 3;
    inds_t[o + 0] = tt;
    inds_t[o + 1] = reflect_i(h + dy, H_);
    inds_t[o + 2] = reflect_i(w + dx, W_);
  }
}

// ---------------- 3) refine: wave32 per (query,shift) row ----------------
// WMMA broadcast mapping: A = 16 candidate patches (vid1), B = query chunk
// replicated into all 16 columns; column 0 of D = 16 candidate scores.
extern "C" __global__ void __launch_bounds__(32)
nls_refine_kernel(const unsigned short* __restrict__ v0,
                  const unsigned short* __restrict__ v1,
                  const int* __restrict__ inds_t,
                  float* __restrict__ dists, int* __restrict__ inds) {
  const int qp = blockIdx.x;        // 0..QS_-1
  const int q  = qp / S_;
  const int lane = threadIdx.x;
  const int r  = lane & 15;
  const bool hi = lane >= 16;

  const int tq = q / (NQ_ * NQ_);
  const int hq = ((q % (NQ_ * NQ_)) / NQ_) * STRIDE0_;
  const int wq = (q % NQ_) * STRIDE0_;

  __shared__ __align__(16) unsigned short Qf[FEAT_];  // query patch, bf16
  __shared__ __align__(16) float diag[256];
  __shared__ float scores[64];
  __shared__ int ct[KK_], ch[KK_], cw[KK_];

  for (int pp = lane; pp < 49; pp += 32) {
    const int py = pp / 7 - PAD_, px = pp % 7 - PAD_;
    const int h = reflect_i(hq + py, H_), w = reflect_i(wq + px, W_);
    const u32x4* s = (const u32x4*)(v0 + (((size_t)tq * H_ + h) * W_ + w) * C_);
    u32x4* d = (u32x4*)(Qf + pp * C_);
#pragma unroll
    for (int p = 0; p < 8; ++p) d[p] = s[p];
  }
  if (lane < KK_) {
    const int b = (qp * KK_ + lane) * 3;
    ct[lane] = inds_t[b]; ch[lane] = inds_t[b + 1]; cw[lane] = inds_t[b + 2];
  }
  __syncthreads();

  const int o0 = hi ? 16 : 0, o1 = hi ? 24 : 8;
  const int bo = hi ? 16 : 0;

  for (int cg = 0; cg < 4; ++cg) {
    int ci = cg * 16 + r; if (ci >= KK_ * R_) ci = 0;   // pad rows -> cand 0
    const int k = ci / R_, rr = ci % R_;
    const int dr = rr / 3 - 1, dc = rr % 3 - 1;
    const int tc = ct[k], hc = ch[k] + dr, wc = cw[k] + dc;
    const unsigned short* base1 = v1 + (size_t)tc * H_ * W_ * C_;

    int coloff[PS_];
#pragma unroll
    for (int p = 0; p < PS_; ++p)
      coloff[p] = reflect_i(wc + p - PAD_, W_) * C_;

    v8f acc0 = {}, acc1 = {};
    for (int py = 0; py < PS_; ++py) {
      const unsigned short* arow =
          base1 + (size_t)reflect_i(hc + py - PAD_, H_) * (W_ * C_);
      const unsigned short* qrow = Qf + (py * 7) * C_;
#pragma unroll
      for (int px = 0; px < PS_; ++px) {
        ABPack a0, b0, a1, b1;
        const unsigned short* ap = arow + coloff[px];   // candidate (global/L2)
        const unsigned short* bp = qrow + px * C_ + bo; // query broadcast (LDS)
        a0.q[0] = ((const u32x4*)(ap + o0))[0];
        a0.q[1] = ((const u32x4*)(ap + o1))[0];
        b0.q[0] = ((const u32x4*)bp)[0];
        b0.q[1] = ((const u32x4*)bp)[1];
        acc0 = WMMA_BF16(a0.v, b0.v, acc0);
        a1.q[0] = ((const u32x4*)(ap + 32 + o0))[0];
        a1.q[1] = ((const u32x4*)(ap + 32 + o1))[0];
        b1.q[0] = ((const u32x4*)(bp + 32))[0];
        b1.q[1] = ((const u32x4*)(bp + 32))[1];
        acc1 = WMMA_BF16(a1.v, b1.v, acc1);
      }
    }
#pragma unroll
    for (int v = 0; v < 8; ++v) diag[lane * 8 + v] = acc0[v] + acc1[v];
    __syncthreads();
    if (lane < 16) {
      const int ci2 = cg * 16 + r;
      if (ci2 < KK_ * R_) {
        // column 0 lives in lane 0 (rows 0-7) and lane 16 (rows 8-15)
        scores[ci2] = (r < 8) ? diag[0 * 8 + r] : diag[16 * 8 + (r - 8)];
      }
    }
    __syncthreads();
  }

  if (lane == 0) {
    for (int i = 0; i < KK_; ++i) {
      float mb = NEG_INF; int mj = 0;
      for (int j = 0; j < KK_ * R_; ++j) {
        const float v = scores[j];
        const bool g = v > mb;          // branchless argmax, first-wins ties
        mb = g ? v : mb;
        mj = g ? j : mj;
      }
      scores[mj] = NEG_INF;
      const int k = mj / R_, rr = mj % R_;
      const int dr = rr / 3 - 1, dc = rr % 3 - 1;
      dists[(Q_ + qp) * KK_ + i] = mb;
      const int b = ((Q_ + qp) * KK_ + i) * 3;
      inds[b + 0] = ct[k];
      inds[b + 1] = reflect_i(ch[k] + dr, H_);
      inds[b + 2] = reflect_i(cw[k] + dc, W_);
    }
  }
}

// ---------------- launcher ----------------
extern "C" void kernel_launch(void* const* d_in, const int* in_sizes, int n_in,
                              void* d_out, int out_size, void* d_ws, size_t ws_size,
                              hipStream_t stream) {
  const float* vid0  = (const float*)d_in[0];
  const float* vid1  = (const float*)d_in[1];
  const float* fflow = (const float*)d_in[2];
  const float* bflow = (const float*)d_in[3];

  float* dists = (float*)d_out;                                    // (Q+QS)*K
  int*   inds  = (int*)((float*)d_out + (size_t)(Q_ + QS_) * KK_); // (Q+QS)*K*3 int32

  const size_t NVID = (size_t)T_ * H_ * W_ * C_;
  unsigned short* v0bf = (unsigned short*)d_ws;
  unsigned short* v1bf = v0bf + NVID;
  int* inds_t = (int*)(v1bf + NVID);                               // QS*K*3 ints

  const int cth = 256;
  const int cbl = (int)((NVID + cth - 1) / cth);
  nls_convert_kernel<<<cbl, cth, 0, stream>>>(vid0, v0bf);
  nls_convert_kernel<<<cbl, cth, 0, stream>>>(vid1, v1bf);

  nls_exact_kernel<<<T_ * 12 * 12, 32, 0, stream>>>(v0bf, v1bf, dists, inds);

  nls_temporal_kernel<<<(Q_ * KK_ + 255) / 256, 256, 0, stream>>>(
      inds, fflow, bflow, inds_t);

  nls_refine_kernel<<<QS_, 32, 0, stream>>>(v0bf, v1bf, inds_t, dists, inds);
}
